// RGCNLayer_68822555951379
// MI455X (gfx1250) — matrix operand order repaired and checked by reading, not compiled, not run on hardware
//
#include <hip/hip_runtime.h>
#include <math.h>

typedef __attribute__((ext_vector_type(16))) __bf16 v16bf;
typedef __attribute__((ext_vector_type(8)))  __bf16 v8bf;
typedef __attribute__((ext_vector_type(8)))  float  v8f;
typedef __attribute__((ext_vector_type(4)))  float  v4f;

#define BB 4
#define NN 2048
#define HH 64
#define RR 4

// ---------------- Kernel 1: h = tanh(LN(Linear(concat(x, feature)))) --------
__global__ __launch_bounds__(64)
void k_prep(const float* __restrict__ x, const float* __restrict__ feat,
            const float* __restrict__ fcw, const float* __restrict__ fcb,
            const float* __restrict__ lng, const float* __restrict__ lnb,
            float* __restrict__ hout) {
  int row = blockIdx.x;           // b*N + n  (8192 rows)
  int t   = threadIdx.x;          // 0..63 (output channel)
  __shared__ float in[66];
  __shared__ float red[64];
  in[t] = x[(size_t)row * HH + t];
  if (t < 2) in[HH + t] = feat[(size_t)row * 2 + t];
  __syncthreads();
  float acc = fcb[t];
  #pragma unroll
  for (int f = 0; f < HH + 2; ++f) acc += in[f] * fcw[t * (HH + 2) + f];
  // LayerNorm over 64 channels (population variance, eps=1e-5)
  red[t] = acc; __syncthreads();
  for (int s = 32; s > 0; s >>= 1) { if (t < s) red[t] += red[t + s]; __syncthreads(); }
  float mu = red[0] * (1.0f / 64.0f);
  __syncthreads();
  float d = acc - mu;
  red[t] = d * d; __syncthreads();
  for (int s = 32; s > 0; s >>= 1) { if (t < s) red[t] += red[t + s]; __syncthreads(); }
  float var = red[0] * (1.0f / 64.0f);
  float v = d * rsqrtf(var + 1e-5f) * lng[t] + lnb[t];
  hout[(size_t)row * HH + t] = tanhf(v);
}

// ---------------- Kernel 2: Wr[r][h][k] = sum_v a[r,v] * W[h,v,k] -----------
__global__ __launch_bounds__(256)
void k_wr(const float* __restrict__ a, const float* __restrict__ W,
          float* __restrict__ wr) {
  int i = blockIdx.x * 256 + threadIdx.x;     // < 5*64*64
  if (i >= 5 * HH * HH) return;
  int k = i & 63, h = (i >> 6) & 63, r = i >> 12;
  float s = 0.f;
  #pragma unroll
  for (int v = 0; v < 8; ++v) s += a[r * 8 + v] * W[(h * 8 + v) * HH + k];
  wr[i] = s;
}

// ---- Kernel 3: out_init = h@Wr[0] + h ; HrT[b][r][k][m] = bf16(h@Wr[r+1]) --
__global__ __launch_bounds__(320)
void k_hr(const float* __restrict__ h, const float* __restrict__ wr,
          float* __restrict__ outp, __bf16* __restrict__ hrt) {
  int row = blockIdx.x;                 // b*N + m
  int b = row / NN, m = row % NN;
  __shared__ float hr_[HH];
  int t = threadIdx.x;                  // 0..319  -> (r = t/64, k = t%64)
  if (t < HH) hr_[t] = h[(size_t)row * HH + t];
  __syncthreads();
  int r = t >> 6, k = t & 63;
  float s = 0.f;
  #pragma unroll
  for (int hh = 0; hh < HH; ++hh) s += hr_[hh] * wr[(r * HH + hh) * HH + k];
  if (r == 0) {
    outp[(size_t)row * HH + k] = s + hr_[k];          // h@Wr[0] + h
  } else {
    hrt[(((size_t)b * RR + (r - 1)) * HH + k) * NN + m] = (__bf16)s;
  }
}

// ---------------- Kernel 4: partial[r][b] = adj[b,r] @ Hr[b,r]  (bf16 WMMA) -
struct ABuf { v4f f0, f1, f2, f3; };          // raw f32 A fragment (16 floats)
struct BBuf { v8bf lo[4], hi[4]; };           // bf16 B fragments, 4 col-groups

__device__ __forceinline__ void loadA(const float* __restrict__ ap, ABuf& A) {
  A.f0 = *(const v4f*)(ap);
  A.f1 = *(const v4f*)(ap + 4);
  A.f2 = *(const v4f*)(ap + 16);
  A.f3 = *(const v4f*)(ap + 20);
}

__device__ __forceinline__ void loadB(const __bf16* __restrict__ bbase,
                                      int l15, int m0, int kbB, BBuf& B) {
  #pragma unroll
  for (int j = 0; j < 4; ++j) {
    const __bf16* bp = bbase + (size_t)(j * 16 + l15) * NN + m0 + kbB;
    B.lo[j] = *(const v8bf*)bp;
    B.hi[j] = *(const v8bf*)(bp + 16);
  }
}

__device__ __forceinline__ void mma_step(const ABuf& A, const BBuf& B,
                                         v8f& c0, v8f& c1, v8f& c2, v8f& c3) {
  v16bf av;
  #pragma unroll
  for (int i = 0; i < 4; ++i) {
    av[i]      = (__bf16)A.f0[i];
    av[4 + i]  = (__bf16)A.f1[i];
    av[8 + i]  = (__bf16)A.f2[i];
    av[12 + i] = (__bf16)A.f3[i];
  }
  v8f* cs[4] = { &c0, &c1, &c2, &c3 };
  #pragma unroll
  for (int j = 0; j < 4; ++j) {
    v16bf bv;
    #pragma unroll
    for (int i = 0; i < 8; ++i) { bv[i] = B.lo[j][i]; bv[8 + i] = B.hi[j][i]; }
    *cs[j] = __builtin_amdgcn_wmma_f32_16x16x32_bf16(
        false, av, false, bv, (short)0, *cs[j], false, false);
  }
}

__global__ __launch_bounds__(128)
void k_gemm(const float* __restrict__ adj, const __bf16* __restrict__ hrt,
            float* __restrict__ part) {
  // grid.x = B * R * (N/64) = 512 ; block = 4 waves, 64-row output tile x 64 cols
  int blk = blockIdx.x;
  int mt  = blk & 31;                 // 64-row tile index
  int r   = (blk >> 5) & (RR - 1);
  int b   = blk >> 7;
  int lane = threadIdx.x & 31;
  int wave = threadIdx.x >> 5;        // 0..3
  int l15  = lane & 15;
  bool hi  = lane >= 16;

  int rowA = mt * 64 + wave * 16 + l15;                       // A row this lane feeds
  const float* arow = adj + (((size_t)(b * RR + r)) * NN + rowA) * NN;
  const int kbA = hi ? 8 : 0;                                  // A-frag K grouping
  const int kbB = hi ? 8 : 0;                                  // B-frag K grouping
  const __bf16* bbase = hrt + ((size_t)(b * RR + r)) * (HH * NN);

  v8f c0 = {}, c1 = {}, c2 = {}, c3 = {};

  // 2-stage software pipeline over K (NN/32 = 64 steps, unrolled by 2)
  ABuf A0, A1; BBuf B0, B1;
  loadA(arow + 0 + kbA, A0);
  loadB(bbase, l15, 0, kbB, B0);
  for (int m0 = 0; m0 < NN; m0 += 64) {
    loadA(arow + (m0 + 32) + kbA, A1);            // prefetch stage n+1
    loadB(bbase, l15, m0 + 32, kbB, B1);
    mma_step(A0, B0, c0, c1, c2, c3);             // compute stage n
    if (m0 + 64 < NN) {
      loadA(arow + (m0 + 64) + kbA, A0);          // prefetch stage n+2
      loadB(bbase, l15, m0 + 64, kbB, B0);
    }
    mma_step(A1, B1, c0, c1, c2, c3);             // compute stage n+1
  }

  // ---- store: D vgpr i -> row (i + hi*8), col (j*16 + l15)
  float* pbase = part + ((size_t)r * BB + b) * ((size_t)NN * HH);
  int rbase = mt * 64 + wave * 16 + (hi ? 8 : 0);
  v8f cv[4] = { c0, c1, c2, c3 };
  #pragma unroll
  for (int j = 0; j < 4; ++j) {
    #pragma unroll
    for (int i = 0; i < 8; ++i) {
      pbase[(size_t)(rbase + i) * HH + j * 16 + l15] = cv[j][i];
    }
  }
}

// ---------------- Kernel 5: out += sum_r partial[r] -------------------------
__global__ __launch_bounds__(256)
void k_red(float* __restrict__ out, const float* __restrict__ part) {
  int i = blockIdx.x * 256 + threadIdx.x;    // < B*N*H = 524288
  float s = out[i];
  #pragma unroll
  for (int r = 0; r < RR; ++r) s += part[(size_t)r * (BB * NN * HH) + i];
  out[i] = s;
}

// ---------------------------------------------------------------------------
extern "C" void kernel_launch(void* const* d_in, const int* in_sizes, int n_in,
                              void* d_out, int out_size, void* d_ws, size_t ws_size,
                              hipStream_t stream) {
  const float* x    = (const float*)d_in[0];   // [B,N,H]
  const float* feat = (const float*)d_in[1];   // [B,N,2]
  const float* adj  = (const float*)d_in[2];   // [B,R,N,N]
  const float* fcw  = (const float*)d_in[3];   // [H,H+2]
  const float* fcb  = (const float*)d_in[4];   // [H]
  const float* lng  = (const float*)d_in[5];   // [H]
  const float* lnb  = (const float*)d_in[6];   // [H]
  const float* W    = (const float*)d_in[7];   // [H,V,H]
  const float* a    = (const float*)d_in[8];   // [R+1,V]
  float* out = (float*)d_out;

  char* ws = (char*)d_ws;
  float*  h    = (float*)(ws);                          // 2 MB   [B,N,H] f32
  float*  wr   = (float*)(ws + 2097152);                // 80 KB  [5,H,H] f32
  __bf16* hrt  = (__bf16*)(ws + 2228224);               // 4 MB   [B,R,H,N] bf16 (transposed Hr)
  float*  part = (float*)(ws + 6422528);                // 8 MB   [R,B,N,H] f32 partials

  k_prep<<<BB * NN, 64, 0, stream>>>(x, feat, fcw, fcb, lng, lnb, h);
  k_wr  <<<(5 * HH * HH + 255) / 256, 256, 0, stream>>>(a, W, wr);
  k_hr  <<<BB * NN, 320, 0, stream>>>(h, wr, out, hrt);
  k_gemm<<<BB * RR * (NN / 64), 128, 0, stream>>>(adj, hrt, part);
  k_red <<<(BB * NN * HH) / 256, 256, 0, stream>>>(out, part);
}
